// DynamicLSTM_90623809945711
// MI455X (gfx1250) — matrix operand order repaired and checked by reading, not compiled, not run on hardware
//
#include <hip/hip_runtime.h>
#include <hip/hip_bf16.h>
#include <math.h>

// ---------------------------------------------------------------------------
// DynamicLSTM on MI455X (gfx1250, wave32, WMMA):
//   - static weights pre-converted to bf16 once per launch
//   - cell GEMM (1024x1024x384) via v_wmma_f32_16x16x32_bf16, all four gates
//     fused per-wave (4 accumulators share one A fragment)
//   - K loop software-pipelined with a rotating 3-slot fragment buffer
//     (prefetch distance 2). The schedule is PINNED with an inline-asm
//     barrier (memory clobber + accumulators as operands) so the backend
//     scheduler cannot sink the prefetch loads back to their uses;
//     amdgpu_waves_per_eu(1,2) unlocks the ~152-VGPR budget this needs.
//   - c/h written straight into a 16-slot ring buffer (the reference's
//     obs_c/obs_h window); LAMDA==1 makes wc/wh a pure ring gather
//   - policy fc1 also WMMA (same pipelining); fc2 + Gumbel sampling + gather
//     in VALU with LDS staging
// ---------------------------------------------------------------------------

typedef __attribute__((ext_vector_type(16))) __bf16 v16bf;
typedef __attribute__((ext_vector_type(8)))  __bf16 v8bf;
typedef __attribute__((ext_vector_type(8)))  float  v8f;

#define WMMA_BF16(a, b, c) \
  __builtin_amdgcn_wmma_f32_16x16x32_bf16(false, (a), false, (b), (short)0, (c), false, false)

// Scheduling pin: loads issued before this cannot sink below it ("memory"
// clobber); WMMAs writing the accumulators cannot hoist above it (acc are
// in/out operands). Compiles to zero instructions.
#define SCHED_PIN4(a0, a1, a2, a3) \
  asm volatile("" : "+v"(a0), "+v"(a1), "+v"(a2), "+v"(a3)::"memory")

constexpr int B  = 1024;
constexpr int T  = 256;
constexpr int D  = 128;
constexpr int H  = 256;
constexpr int U  = 64;
constexpr int O  = 128;
constexpr int DG = D + H;   // 384 = concat width, K dim of both big GEMMs
constexpr int G4 = 4 * H;   // 1024 gate columns
constexpr int KT = DG / 32; // 12 K-steps of 32

__device__ __forceinline__ float sigmoidf_(float x) { return 1.0f / (1.0f + __expf(-x)); }

__device__ __forceinline__ uint32_t hash_u32(uint32_t x) {
  x ^= x >> 16; x *= 0x7feb352du;
  x ^= x >> 15; x *= 0x846ca68bu;
  x ^= x >> 16;
  return x;
}

// A fragment (16x32 bf16, MxK) from a bf16 row-major matrix with row stride DG.
// ISA layout: lanes 0-15 row M=lane hold K {0..7,16..23}; lanes 16-31 hold
// K {8..15,24..31} for the same rows.
__device__ __forceinline__ v16bf load_a_bf16(const __bf16* __restrict__ base,
                                             int m0, int k0, int lane) {
  const int row  = m0 + (lane & 15);
  const int koff = k0 + ((lane >> 4) << 3);        // +0 or +8
  const __bf16* p = base + (size_t)row * DG + koff;
  v8bf lo = *(const v8bf*)(p);                      // K koff..koff+7
  v8bf hi = *(const v8bf*)(p + 16);                 // K koff+16..koff+23
  v16bf f;
#pragma unroll
  for (int e = 0; e < 8; ++e) { f[e] = lo[e]; f[e + 8] = hi[e]; }
  return f;
}

// B fragment (32x16 bf16, KxN) taken from weight matrix W[n][k] (row-major,
// row stride DG) so B[k][n] = W[n][k]. Lanes 0-15: col n, K {0..7,16..23}; lanes
// 16-31: same cols, K {8..15,24..31} (two b128 loads per lane).
__device__ __forceinline__ v16bf load_b_bf16(const __bf16* __restrict__ w,
                                             int n0, int k0, int lane) {
  const int col = n0 + (lane & 15);
  const int kb  = k0 + ((lane >> 4) << 3);          // +0 or +8
  const __bf16* p = w + (size_t)col * DG + kb;
  v8bf lo = *(const v8bf*)(p);
  v8bf hi = *(const v8bf*)(p + 16);
  v16bf f;
#pragma unroll
  for (int e = 0; e < 8; ++e) { f[e] = lo[e]; f[e + 8] = hi[e]; }
  return f;
}

// A fragment built on the fly from f32 obs = [x_t | h] (policy GEMM).
__device__ __forceinline__ v16bf load_a_obs(const float* __restrict__ xrow,
                                            const float* __restrict__ hrow,
                                            int k0, int lane) {
  const int koff = k0 + ((lane >> 4) << 3);
  v16bf f;
#pragma unroll
  for (int c = 0; c < 2; ++c) {
    const int k = koff + c * 16;
    const float* p = (k < D) ? (xrow + k) : (hrow + (k - D));
#pragma unroll
    for (int e = 0; e < 8; ++e) f[c * 8 + e] = (__bf16)p[e];
  }
  return f;
}

// ---------------------------------------------------------------------------
// One-time weight conversion: Wcat = [x2h_w | h2h_w] (1024x384) and fc1_w
// (64x384) to bf16.
// ---------------------------------------------------------------------------
__global__ void convert_kernel(const float* __restrict__ x2h_w,
                               const float* __restrict__ h2h_w,
                               const float* __restrict__ fc1_w,
                               __bf16* __restrict__ wcat,
                               __bf16* __restrict__ fc1wb) {
  const int idx = blockIdx.x * blockDim.x + threadIdx.x;
  if (idx < G4 * DG) {
    const int j = idx / DG, k = idx % DG;
    wcat[idx] = (__bf16)((k < D) ? x2h_w[j * D + k] : h2h_w[j * H + (k - D)]);
  } else {
    const int i2 = idx - G4 * DG;
    if (i2 < U * DG) fc1wb[i2] = (__bf16)fc1_w[i2];
  }
}

// Initial state: cat0 = [bf16(x[:,0,:]) | 0], wc = 0.
__global__ void init_kernel(const float* __restrict__ input,
                            __bf16* __restrict__ cat,
                            float* __restrict__ wc) {
  const int idx = blockIdx.x * blockDim.x + threadIdx.x;
  if (idx < B * DG) {
    const int b = idx / DG, k = idx % DG;
    cat[idx] = (k < D) ? (__bf16)input[(size_t)b * T * D + k] : (__bf16)0.0f;
  } else {
    const int i2 = idx - B * DG;
    if (i2 < B * H) wc[i2] = 0.0f;
  }
}

// ---------------------------------------------------------------------------
// Fused LSTM cell: gates = cat(1024x384,bf16) @ Wcat^T; each wave owns a
// 16(row)x16(hidden-col) tile with 4 f32 accumulators (i,f,g,o).
// Rotating 3-slot fragment pipeline, prefetch distance 2, schedule pinned.
// 1024 waves = 128 blocks x 8 waves.
// ---------------------------------------------------------------------------
__global__ __launch_bounds__(256)
__attribute__((amdgpu_waves_per_eu(1, 2)))
void cell_kernel(
    const __bf16* __restrict__ cat, const __bf16* __restrict__ wcat,
    const float* __restrict__ x2h_b, const float* __restrict__ h2h_b,
    const float* __restrict__ wc,
    float* __restrict__ ring_c, float* __restrict__ ring_h, int slot) {
  const int tid   = threadIdx.x;
  const int lane  = tid & 31;
  const int gwave = blockIdx.x * 8 + (tid >> 5);
  const int m0    = (gwave >> 4) * 16;   // 64 row tiles
  const int h0    = (gwave & 15) * 16;   // 16 hidden-col tiles

  v8f acc0 = {}, acc1 = {}, acc2 = {}, acc3 = {};

  v16bf a_buf[3];
  v16bf b_buf[3][4];
#pragma unroll
  for (int s = 0; s < 2; ++s) {          // prologue: k-steps 0 and 1 in flight
    a_buf[s] = load_a_bf16(cat, m0, s * 32, lane);
#pragma unroll
    for (int g = 0; g < 4; ++g) b_buf[s][g] = load_b_bf16(wcat, g * H + h0, s * 32, lane);
  }

#pragma unroll
  for (int kt = 0; kt < KT; ++kt) {
    if (kt + 2 < KT) {
      const int sl = (kt + 2) % 3;
      const int kn = (kt + 2) * 32;
      a_buf[sl] = load_a_bf16(cat, m0, kn, lane);
#pragma unroll
      for (int g = 0; g < 4; ++g) b_buf[sl][g] = load_b_bf16(wcat, g * H + h0, kn, lane);
    }
    SCHED_PIN4(acc0, acc1, acc2, acc3);  // loads stay above, WMMAs stay below
    const int sc = kt % 3;
    acc0 = WMMA_BF16(a_buf[sc], b_buf[sc][0], acc0);
    acc1 = WMMA_BF16(a_buf[sc], b_buf[sc][1], acc1);
    acc2 = WMMA_BF16(a_buf[sc], b_buf[sc][2], acc2);
    acc3 = WMMA_BF16(a_buf[sc], b_buf[sc][3], acc3);
  }

  const int colh = h0 + (lane & 15);
  const float bi = x2h_b[colh]         + h2h_b[colh];
  const float bf = x2h_b[H + colh]     + h2h_b[H + colh];
  const float bg = x2h_b[2 * H + colh] + h2h_b[2 * H + colh];
  const float bo = x2h_b[3 * H + colh] + h2h_b[3 * H + colh];

  float* __restrict__ cdst = ring_c + (size_t)slot * B * H;
  float* __restrict__ hdst = ring_h + (size_t)slot * B * H;
#pragma unroll
  for (int r = 0; r < 8; ++r) {
    const int row = m0 + r + ((lane >> 4) << 3);     // C layout: M=r / r+8
    const float iv = sigmoidf_(acc0[r] + bi);
    const float fv = sigmoidf_(acc1[r] + bf);
    const float gv = tanhf(acc2[r] + bg);
    const float ov = sigmoidf_(acc3[r] + bo);
    const float cp = wc[(size_t)row * H + colh];
    const float cn = cp * fv + iv * gv;
    cdst[(size_t)row * H + colh] = cn;
    hdst[(size_t)row * H + colh] = ov * tanhf(cn);
  }
}

// ---------------------------------------------------------------------------
// Policy step j: fc1 (WMMA bf16, pinned pipeline) -> fc2 + Gumbel sampling ->
// ring gather. 8 blocks x 256 threads; each block handles 128 batch rows.
// Writes agent_action/agent_prob, builds next cat(bf16) and wc(f32).
// ---------------------------------------------------------------------------
__global__ __launch_bounds__(256)
__attribute__((amdgpu_waves_per_eu(1, 2)))
void policy_kernel(
    const float* __restrict__ input, const __bf16* __restrict__ fc1w,
    const float* __restrict__ fc1_b, const float* __restrict__ fc2_w,
    const float* __restrict__ fc2_b,
    const float* __restrict__ ring_c, const float* __restrict__ ring_h,
    __bf16* __restrict__ cat, float* __restrict__ wc,
    float* __restrict__ out_act, float* __restrict__ out_logp, int step) {
  __shared__ float u_lds[128][U + 1];
  __shared__ int   act_lds[128];

  const int tid   = threadIdx.x;
  const int lane  = tid & 31;
  const int wave  = tid >> 5;
  const int t     = step + 1;                       // x index
  const int hslot = step & 15;                      // current h in ring
  const float* __restrict__ hbase = ring_h + (size_t)hslot * B * H;

  // --- phase 1: u = obs @ fc1_w^T + fc1_b via WMMA (16 rows x 64 cols/wave)
  {
    const int m0   = blockIdx.x * 128 + wave * 16;
    const int mrow = m0 + (lane & 15);
    const float* xrow = input + ((size_t)mrow * T + t) * D;
    const float* hrow = hbase + (size_t)mrow * H;
    v8f acc0 = {}, acc1 = {}, acc2 = {}, acc3 = {};

    v16bf a_buf[3];
    v16bf b_buf[3][4];
#pragma unroll
    for (int s = 0; s < 2; ++s) {
      a_buf[s] = load_a_obs(xrow, hrow, s * 32, lane);
#pragma unroll
      for (int nt = 0; nt < 4; ++nt) b_buf[s][nt] = load_b_bf16(fc1w, nt * 16, s * 32, lane);
    }

#pragma unroll
    for (int kt = 0; kt < KT; ++kt) {
      if (kt + 2 < KT) {
        const int sl = (kt + 2) % 3;
        const int kn = (kt + 2) * 32;
        a_buf[sl] = load_a_obs(xrow, hrow, kn, lane);
#pragma unroll
        for (int nt = 0; nt < 4; ++nt) b_buf[sl][nt] = load_b_bf16(fc1w, nt * 16, kn, lane);
      }
      SCHED_PIN4(acc0, acc1, acc2, acc3);
      const int sc = kt % 3;
      acc0 = WMMA_BF16(a_buf[sc], b_buf[sc][0], acc0);
      acc1 = WMMA_BF16(a_buf[sc], b_buf[sc][1], acc1);
      acc2 = WMMA_BF16(a_buf[sc], b_buf[sc][2], acc2);
      acc3 = WMMA_BF16(a_buf[sc], b_buf[sc][3], acc3);
    }
    v8f accv[4] = {acc0, acc1, acc2, acc3};
#pragma unroll
    for (int nt = 0; nt < 4; ++nt) {
      const int col  = nt * 16 + (lane & 15);
      const float bb = fc1_b[col];
#pragma unroll
      for (int r = 0; r < 8; ++r) {
        const int rl = wave * 16 + r + ((lane >> 4) << 3);
        u_lds[rl][col] = accv[nt][r] + bb;
      }
    }
  }
  __syncthreads();

  // --- phase 2: fc2 logits + deterministic Gumbel-max sample + logp
  if (tid < 128) {
    const int rl = tid;
    const int rg = blockIdx.x * 128 + rl;
    float lg[16];
    float m = -1e30f;
#pragma unroll
    for (int a = 0; a < 16; ++a) {
      float s = fc2_b[a];
      for (int u = 0; u < U; ++u) s += u_lds[rl][u] * fc2_w[a * U + u];
      lg[a] = s;
      m = fmaxf(m, s);
    }
    float lse = 0.0f;
#pragma unroll
    for (int a = 0; a < 16; ++a) lse += __expf(lg[a] - m);
    lse = __logf(lse);
    int best = 0; float bestv = -1e30f;
#pragma unroll
    for (int a = 0; a < 16; ++a) {
      const uint32_t h = hash_u32(0x9E3779B9u * (uint32_t)step ^
                                  0x85EBCA6Bu * (uint32_t)rg ^
                                  0xC2B2AE35u * (uint32_t)a);
      const float uu  = (float)(h >> 8) * (1.0f / 16777216.0f) + 1e-20f;
      const float gum = -__logf(-__logf(uu) + 1e-20f);
      const float v   = lg[a] + gum;
      if (v > bestv) { bestv = v; best = a; }
    }
    act_lds[rl]             = best;
    out_act [step * B + rg] = (float)best;
    out_logp[step * B + rg] = lg[best] - m - lse;
  }
  __syncthreads();

  // --- phase 3: gather wc/wh from ring (LAMDA==1) + build next cat (bf16)
  {
    const int rl   = tid >> 1;
    const int half = tid & 1;
    const int rg   = blockIdx.x * 128 + rl;
    const int a    = act_lds[rl];
    const int src  = step - (15 - a);               // push step of selected obs
    const bool valid = (src >= 0);
    const size_t roff = (size_t)rg * H;
    const float* __restrict__ csrc = ring_c + (size_t)(src & 15) * B * H + roff;
    const float* __restrict__ hsrc = ring_h + (size_t)(src & 15) * B * H + roff;
    __bf16* __restrict__ catrow = cat + (size_t)rg * DG;
    for (int k = half * 128; k < half * 128 + 128; ++k) {
      const float cv = valid ? csrc[k] : 0.0f;
      const float hv = valid ? hsrc[k] : 0.0f;
      wc[roff + k]  = cv;
      catrow[D + k] = (__bf16)hv;
    }
    const float* xrow = input + ((size_t)rg * T + t) * D;
    for (int k = half * 64; k < half * 64 + 64; ++k) catrow[k] = (__bf16)xrow[k];
  }
}

// ---------------------------------------------------------------------------
// Epilogue: opt = softmax(h_final @ out_w^T + out_b). One block per row.
// ---------------------------------------------------------------------------
__global__ __launch_bounds__(128) void out_kernel(
    const float* __restrict__ ring_h, const float* __restrict__ out_w,
    const float* __restrict__ out_b, float* __restrict__ opt) {
  __shared__ float red[128];
  const int row = blockIdx.x;
  const int o   = threadIdx.x;
  const float* hrow = ring_h + (size_t)15 * B * H + (size_t)row * H;  // slot (T-1)&15
  float s = out_b[o];
  for (int k = 0; k < H; ++k) s += hrow[k] * out_w[o * H + k];
  red[o] = s; __syncthreads();
  for (int st = 64; st > 0; st >>= 1) {
    if (o < st) red[o] = fmaxf(red[o], red[o + st]);
    __syncthreads();
  }
  const float m = red[0]; __syncthreads();
  const float e = __expf(s - m);
  red[o] = e; __syncthreads();
  for (int st = 64; st > 0; st >>= 1) {
    if (o < st) red[o] += red[o + st];
    __syncthreads();
  }
  opt[(size_t)row * O + o] = e / red[0];
}

// ---------------------------------------------------------------------------
extern "C" void kernel_launch(void* const* d_in, const int* in_sizes, int n_in,
                              void* d_out, int out_size, void* d_ws, size_t ws_size,
                              hipStream_t stream) {
  (void)in_sizes; (void)n_in; (void)out_size; (void)ws_size;

  const float* input = (const float*)d_in[0];
  const float* fc1_w = (const float*)d_in[1];
  const float* fc1_b = (const float*)d_in[2];
  const float* fc2_w = (const float*)d_in[3];
  const float* fc2_b = (const float*)d_in[4];
  const float* x2h_w = (const float*)d_in[5];
  const float* x2h_b = (const float*)d_in[6];
  const float* h2h_w = (const float*)d_in[7];
  const float* h2h_b = (const float*)d_in[8];
  const float* out_w = (const float*)d_in[9];
  const float* out_b = (const float*)d_in[10];
  float* out = (float*)d_out;

  // Workspace layout (bytes), all 256-aligned:
  char* ws = (char*)d_ws;
  __bf16* wcat  = (__bf16*)(ws);              //  786,432  Wcat bf16 (1024x384)
  __bf16* fc1wb = (__bf16*)(ws + 786432);     //   49,152  fc1_w bf16 (64x384)
  __bf16* cat   = (__bf16*)(ws + 835584);     //  786,432  [x_t|wh] bf16 (1024x384)
  float*  wc    = (float*) (ws + 1622016);    // 1,048,576 gathered c (1024x256)
  float*  ringc = (float*) (ws + 2670592);    // 16 MB ring of c (16x1024x256)
  float*  ringh = (float*) (ws + 19447808);   // 16 MB ring of h

  float* out_opt  = out;                      // (1024,128)
  float* out_act  = out + B * O;              // (255,1024)
  float* out_logp = out + B * O + (T - 1) * B;// (255,1024)

  convert_kernel<<<(G4 * DG + U * DG + 255) / 256, 256, 0, stream>>>(
      x2h_w, h2h_w, fc1_w, wcat, fc1wb);
  init_kernel<<<(B * DG + B * H + 255) / 256, 256, 0, stream>>>(input, cat, wc);

  // Initial cell (x=input[:,0], wc=wh=0) -> ring slot 0.
  cell_kernel<<<128, 256, 0, stream>>>(cat, wcat, x2h_b, h2h_b, wc, ringc, ringh, 0);

  // 255 sequential steps: policy (sample+gather) then cell -> slot (j+1)&15.
  for (int j = 0; j < T - 1; ++j) {
    policy_kernel<<<8, 256, 0, stream>>>(input, fc1wb, fc1_b, fc2_w, fc2_b,
                                         ringc, ringh, cat, wc,
                                         out_act, out_logp, j);
    cell_kernel<<<128, 256, 0, stream>>>(cat, wcat, x2h_b, h2h_b, wc,
                                         ringc, ringh, (j + 1) & 15);
  }

  out_kernel<<<B, 128, 0, stream>>>(ringh, out_w, out_b, out_opt);
}
